// IM_GCN_43911745634335
// MI455X (gfx1250) — matrix operand order repaired and checked by reading, compiled, not run on hardware
//
#include <hip/hip_runtime.h>
#include <hip/hip_bf16.h>
#include <stdint.h>

// ---------------------------------------------------------------------------
// IM-GCN non-local block for MI455X (gfx1250), f16 WMMA + fp32 accumulate.
// ---------------------------------------------------------------------------

typedef __attribute__((ext_vector_type(16))) _Float16 v16h;
typedef __attribute__((ext_vector_type(8)))  _Float16 v8h;
typedef __attribute__((ext_vector_type(8)))  float    v8f;

#define BB 8
#define DD 256
#define CC 256
#define NN 2048

__constant__ const float kBnEps = 1e-5f;

__device__ __forceinline__ v8f vzero8() {
    v8f z = {0.f, 0.f, 0.f, 0.f, 0.f, 0.f, 0.f, 0.f};
    return z;
}

__device__ __forceinline__ v16h cat16(v8h lo, v8h hi) {
    return __builtin_shufflevector(lo, hi, 0, 1, 2, 3, 4, 5, 6, 7,
                                           8, 9, 10, 11, 12, 13, 14, 15);
}

// A-operand (16x32 f16, M rows x K): lane -> row = lane&15,
// K chunks [(lane>>4)*8, +8) in VGPR0-3 and +16 in VGPR4-7 (ISA 7.12.2).
__device__ __forceinline__ v16h load_aop(const _Float16* base, int ld, int lane) {
    const _Float16* p = base + (lane & 15) * ld + ((lane >> 4) << 3);
    v8h h0 = *(const v8h*)p;
    v8h h1 = *(const v8h*)(p + 16);
    return cat16(h0, h1);
}

// B-operand (32x16 f16, K x N cols), B stored as (cols, K) row-major:
// lane -> col = lane&15, K chunks [(lane>>4)*8, +8) and +16 (same as A-operand).
__device__ __forceinline__ v16h load_bop(const _Float16* base, int ld, int lane) {
    const _Float16* p = base + (lane & 15) * ld + ((lane >> 4) << 3);
    v8h h0 = *(const v8h*)p;
    v8h h1 = *(const v8h*)(p + 16);
    return cat16(h0, h1);
}

__device__ __forceinline__ v8f wmma16(v16h a, v16h b, v8f c) {
    return __builtin_amdgcn_wmma_f32_16x16x32_f16(false, a, false, b,
                                                  (short)0, c, false, false);
}

// ---------------------------------------------------------------------------
// K1: fuse fc1*theta -> Aw, fc2*phi -> Bw, cast g_w / W_w to f16, fused biases.
// grid (256, 4) x 256 threads.
// ---------------------------------------------------------------------------
__global__ void k_prep(const float* __restrict__ fc1_w, const float* __restrict__ theta_w,
                       const float* __restrict__ theta_b, const float* __restrict__ fc1_b,
                       const float* __restrict__ fc2_w, const float* __restrict__ phi_w,
                       const float* __restrict__ phi_b, const float* __restrict__ fc2_b,
                       const float* __restrict__ g_w, const float* __restrict__ W_w,
                       _Float16* __restrict__ Aw, _Float16* __restrict__ Bw,
                       _Float16* __restrict__ Gw, _Float16* __restrict__ Ww,
                       float* __restrict__ Abias, float* __restrict__ Bbias) {
    int d = threadIdx.x;
    int c = blockIdx.x;
    int task = blockIdx.y;
    if (task == 0) {
        float s = 0.f;
        for (int k = 0; k < CC; ++k) s = fmaf(fc1_w[c * CC + k], theta_w[k * DD + d], s);
        Aw[c * DD + d] = (_Float16)s;
        if (d == 0) {
            float sb = 0.f;
            for (int k = 0; k < CC; ++k) sb = fmaf(fc1_w[c * CC + k], theta_b[k], sb);
            Abias[c] = sb + fc1_b[c];
        }
    } else if (task == 1) {
        float s = 0.f;
        for (int k = 0; k < CC; ++k) s = fmaf(fc2_w[c * CC + k], phi_w[k * DD + d], s);
        Bw[c * DD + d] = (_Float16)s;
        if (d == 0) {
            float sb = 0.f;
            for (int k = 0; k < CC; ++k) sb = fmaf(fc2_w[c * CC + k], phi_b[k], sb);
            Bbias[c] = sb + fc2_b[c];
        }
    } else if (task == 2) {
        Gw[c * DD + d] = (_Float16)g_w[c * DD + d];
    } else {
        Ww[c * CC + d] = (_Float16)W_w[c * CC + d];  // W_w is (D, C) row-major
    }
}

// ---------------------------------------------------------------------------
// K2: v (B, D, N) fp32 -> vT (B, N, D) f16 via LDS tiles.
// grid (N/16, D/16, B), block (16, 16).
// ---------------------------------------------------------------------------
__global__ void k_transpose_v(const float* __restrict__ v, _Float16* __restrict__ vT) {
    __shared__ float tile[16][17];
    int b = blockIdx.z;
    int n0 = blockIdx.x * 16, d0 = blockIdx.y * 16;
    int tx = threadIdx.x, ty = threadIdx.y;
    tile[ty][tx] = v[((long long)b * DD + d0 + ty) * NN + n0 + tx];
    __syncthreads();
    vT[((long long)b * NN + n0 + ty) * DD + d0 + tx] = (_Float16)tile[tx][ty];
}

// ---------------------------------------------------------------------------
// K3: generic WMMA projection GEMM: OUT[M x P] = A[M x K] * Bm[P x K]^T + bias.
// One wave computes a 16 x 64 output tile. All dims multiples of 16/64.
// ---------------------------------------------------------------------------
__global__ void k_proj(const _Float16* __restrict__ A, long long aStride,
                       const _Float16* __restrict__ Bm, long long bStride,
                       _Float16* __restrict__ OUT, long long oStride,
                       const float* __restrict__ rowBias, const float* __restrict__ colBias,
                       int M, int K, int P) {
    int lane = threadIdx.x & 31;
    int wave = (blockIdx.x * blockDim.x + threadIdx.x) >> 5;
    int tilesP = P >> 6;
    int wavesPerB = (M >> 4) * tilesP;
    int b = wave / wavesPerB;
    int rem = wave - b * wavesPerB;
    int rt = rem / tilesP;
    int ct = rem - rt * tilesP;

    const _Float16* Abase = A + (long long)b * aStride + (long long)(rt * 16) * K;
    const _Float16* Bbase = Bm + (long long)b * bStride + (long long)(ct * 64) * K;

    v8f acc[4];
#pragma unroll
    for (int j = 0; j < 4; ++j) acc[j] = vzero8();

    for (int kk = 0; kk < K; kk += 32) {
        v16h av = load_aop(Abase + kk, K, lane);
#pragma unroll
        for (int j = 0; j < 4; ++j) {
            v16h bv = load_bop(Bbase + (long long)(j * 16) * K + kk, K, lane);
            acc[j] = wmma16(av, bv, acc[j]);
        }
    }

    int hi = lane >> 4, c0 = lane & 15;
#pragma unroll
    for (int j = 0; j < 4; ++j) {
        int col = ct * 64 + j * 16 + c0;
        float cb = colBias ? colBias[col] : 0.f;
#pragma unroll
        for (int r = 0; r < 8; ++r) {
            int row = rt * 16 + r + 8 * hi;
            float rb = rowBias ? rowBias[row] : 0.f;
            OUT[(long long)b * oStride + (long long)row * P + col] =
                (_Float16)(acc[j][r] + rb + cb);
        }
    }
}

// ---------------------------------------------------------------------------
// K4: fused attention + W conv. One wave owns a 16-row n-strip:
//   Y[16 x 256] = sum_m (a[16,:] * bm[m-chunk,:]^T / N) * g[m-chunk, :]
//   wy[16 x 256] = Y * Ww^T  -> stored (B, N, D) fp32.
// R tiles are re-laid-out via per-wave LDS scratch (ds_store_b16/ds_load_b128).
// 256 blocks x 128 threads (4 waves/block).
// ---------------------------------------------------------------------------
__global__ void k_fused(const _Float16* __restrict__ a_f, const _Float16* __restrict__ bm_f,
                        const _Float16* __restrict__ g_f, const _Float16* __restrict__ Ww,
                        float* __restrict__ wy) {
    __shared__ _Float16 lds[4][16 * 32];
    const float inv_n = 1.0f / (float)NN;

    int lane = threadIdx.x & 31;
    int wslot = threadIdx.x >> 5;
    int wave = (blockIdx.x * blockDim.x + threadIdx.x) >> 5;
    int b = wave >> 7;     // 128 n-tiles per batch
    int nt = wave & 127;

    const _Float16* aB = a_f + ((long long)b * NN + nt * 16) * CC;
    const _Float16* bmB = bm_f + (long long)b * NN * CC;
    const _Float16* gB = g_f + (long long)b * CC * NN;
    _Float16* myl = &lds[wslot][0];

    int hi = lane >> 4, c0 = lane & 15;
    int arow = lane & 15, akb = (lane >> 4) << 3;

    // Cache the wave's 16 x 256 'a' strip as 8 ready-to-issue A-operand tiles.
    v16h aop[8];
#pragma unroll
    for (int kc = 0; kc < 8; ++kc) aop[kc] = load_aop(aB + kc * 32, CC, lane);

    v8f acc[16];
#pragma unroll
    for (int t = 0; t < 16; ++t) acc[t] = vzero8();

    for (int mm = 0; mm < NN; mm += 32) {
        if (mm + 32 < NN) {
            __builtin_prefetch(bmB + (long long)(mm + 32) * CC, 0, 1);
            __builtin_prefetch(bmB + (long long)(mm + 48) * CC, 0, 1);
        }
        // Two 16x16 R tiles (cols mm..mm+31), K = C = 256.
#pragma unroll
        for (int t = 0; t < 2; ++t) {
            v8f r = vzero8();
#pragma unroll
            for (int kc = 0; kc < 8; ++kc) {
                v16h bv = load_bop(bmB + (long long)(mm + t * 16) * CC + kc * 32, CC, lane);
                r = wmma16(aop[kc], bv, r);
            }
            // scale by 1/N, convert to f16, stage in LDS (row-major 16x32)
#pragma unroll
            for (int rr = 0; rr < 8; ++rr)
                myl[(rr + 8 * hi) * 32 + t * 16 + c0] = (_Float16)(r[rr] * inv_n);
        }
        // Re-load R in A-operand layout (ds_load_b128 x2 per lane).
        v8h r0 = *(const v8h*)&myl[arow * 32 + akb];
        v8h r1 = *(const v8h*)&myl[arow * 32 + akb + 16];
        v16h rop = cat16(r0, r1);

        // Y[:, ct-tile] += R(16x32) * g(32x16)
#pragma unroll
        for (int ctile = 0; ctile < 16; ++ctile) {
            v16h gv = load_bop(gB + (long long)(ctile * 16) * NN + mm, NN, lane);
            acc[ctile] = wmma16(rop, gv, acc[ctile]);
        }
    }

    // Convert Y strip to f16 A-operand tiles through the same LDS scratch.
    v16h yop[8];
#pragma unroll
    for (int kc = 0; kc < 8; ++kc) {
#pragma unroll
        for (int t = 0; t < 2; ++t) {
            v8f y = acc[kc * 2 + t];
#pragma unroll
            for (int rr = 0; rr < 8; ++rr)
                myl[(rr + 8 * hi) * 32 + t * 16 + c0] = (_Float16)y[rr];
        }
        v8h y0 = *(const v8h*)&myl[arow * 32 + akb];
        v8h y1 = *(const v8h*)&myl[arow * 32 + akb + 16];
        yop[kc] = cat16(y0, y1);
    }

    // wy[n, d] = sum_c Y[n, c] * W_w[d, c]   (W_b dropped: cancels under BN)
#pragma unroll
    for (int dt = 0; dt < 16; ++dt) {
        v8f w = vzero8();
#pragma unroll
        for (int kc = 0; kc < 8; ++kc) {
            v16h wv = load_bop(Ww + (long long)(dt * 16) * CC + kc * 32, CC, lane);
            w = wmma16(yop[kc], wv, w);
        }
#pragma unroll
        for (int rr = 0; rr < 8; ++rr) {
            int row = nt * 16 + rr + 8 * hi;
            wy[((long long)b * NN + row) * DD + dt * 16 + c0] = w[rr];
        }
    }
}

// ---------------------------------------------------------------------------
// K5: deterministic per-channel BN statistics over wy (B, N, D).
// grid (D/64), block (64, 4): tx = channel within group (coalesced), ty strides rows.
// ---------------------------------------------------------------------------
__global__ void k_bnstats(const float* __restrict__ wy,
                          float* __restrict__ bnsum, float* __restrict__ bnsumsq) {
    __shared__ float ssum[4][64];
    __shared__ float ssq[4][64];
    int tx = threadIdx.x, ty = threadIdx.y;
    int d = blockIdx.x * 64 + tx;
    float s = 0.f, s2 = 0.f;
    const int rows = BB * NN;
    for (int i = ty; i < rows; i += 4) {
        float x = wy[(long long)i * DD + d];
        s += x;
        s2 = fmaf(x, x, s2);
    }
    ssum[ty][tx] = s;
    ssq[ty][tx] = s2;
    __syncthreads();
    if (ty == 0) {
        float t1 = ssum[0][tx] + ssum[1][tx] + ssum[2][tx] + ssum[3][tx];
        float t2 = ssq[0][tx] + ssq[1][tx] + ssq[2][tx] + ssq[3][tx];
        bnsum[d] = t1;
        bnsumsq[d] = t2;
    }
}

// ---------------------------------------------------------------------------
// K6: BN normalize + residual, with LDS-tiled transpose (B,N,D)->(B,D,N).
// grid (N/16, D/16, B), block (16, 16).
// ---------------------------------------------------------------------------
__global__ void k_final(const float* __restrict__ wy, const float* __restrict__ v,
                        const float* __restrict__ bnsum, const float* __restrict__ bnsumsq,
                        const float* __restrict__ gamma, const float* __restrict__ beta,
                        float* __restrict__ out) {
    __shared__ float tile[16][17];
    const float inv_bn = 1.0f / (float)(BB * NN);
    int b = blockIdx.z;
    int n0 = blockIdx.x * 16, d0 = blockIdx.y * 16;
    int tx = threadIdx.x, ty = threadIdx.y;
    tile[ty][tx] = wy[((long long)b * NN + n0 + ty) * DD + d0 + tx];
    __syncthreads();
    int d = d0 + ty, n = n0 + tx;
    float mean = bnsum[d] * inv_bn;
    float var = bnsumsq[d] * inv_bn - mean * mean;
    float inv = rsqrtf(var + kBnEps);
    long long oidx = ((long long)b * DD + d) * NN + n;
    float val = (tile[tx][ty] - mean) * inv * gamma[d] + beta[d];
    out[oidx] = val + v[oidx];
}

// ---------------------------------------------------------------------------
extern "C" void kernel_launch(void* const* d_in, const int* in_sizes, int n_in,
                              void* d_out, int out_size, void* d_ws, size_t ws_size,
                              hipStream_t stream) {
    (void)in_sizes; (void)n_in; (void)out_size; (void)ws_size;

    const float* v        = (const float*)d_in[0];
    const float* g_w      = (const float*)d_in[1];
    const float* g_b      = (const float*)d_in[2];
    const float* theta_w  = (const float*)d_in[3];
    const float* theta_b  = (const float*)d_in[4];
    const float* phi_w    = (const float*)d_in[5];
    const float* phi_b    = (const float*)d_in[6];
    const float* fc1_w    = (const float*)d_in[7];
    const float* fc1_b    = (const float*)d_in[8];
    const float* fc2_w    = (const float*)d_in[9];
    const float* fc2_b    = (const float*)d_in[10];
    const float* W_w      = (const float*)d_in[11];
    /* d_in[12] = W_b: unused — cancels exactly under BatchNorm mean-subtraction */
    const float* bn_gamma = (const float*)d_in[13];
    const float* bn_beta  = (const float*)d_in[14];

    char* ws = (char*)d_ws;
    size_t off = 0;
    auto alloc = [&](size_t bytes) -> char* {
        char* p = ws + off;
        off = (off + bytes + 255) & ~(size_t)255;
        return p;
    };

    _Float16* vT   = (_Float16*)alloc((size_t)BB * NN * DD * 2);
    _Float16* a_f  = (_Float16*)alloc((size_t)BB * NN * CC * 2);
    _Float16* bm_f = (_Float16*)alloc((size_t)BB * NN * CC * 2);
    _Float16* g_f  = (_Float16*)alloc((size_t)BB * CC * NN * 2);
    _Float16* Aw   = (_Float16*)alloc((size_t)CC * DD * 2);
    _Float16* Bw   = (_Float16*)alloc((size_t)CC * DD * 2);
    _Float16* Gw   = (_Float16*)alloc((size_t)CC * DD * 2);
    _Float16* Wwf  = (_Float16*)alloc((size_t)DD * CC * 2);
    float* Abias   = (float*)alloc(CC * 4);
    float* Bbias   = (float*)alloc(CC * 4);
    float* bnsum   = (float*)alloc(DD * 4);
    float* bnsumsq = (float*)alloc(DD * 4);
    float* wy      = (float*)alloc((size_t)BB * NN * DD * 4);

    // 1) fused weights / casts
    k_prep<<<dim3(CC, 4), 256, 0, stream>>>(fc1_w, theta_w, theta_b, fc1_b,
                                            fc2_w, phi_w, phi_b, fc2_b,
                                            g_w, W_w, Aw, Bw, Gw, Wwf, Abias, Bbias);
    // 2) v -> vT (f16)
    k_transpose_v<<<dim3(NN / 16, DD / 16, BB), dim3(16, 16), 0, stream>>>(v, vT);

    // 3) projections (WMMA): a, bm in (B,N,C); g in (B,C,N)
    {
        int waves = BB * (NN / 16) * (CC / 64);           // 4096
        k_proj<<<waves / 4, 128, 0, stream>>>(vT, (long long)NN * DD, Aw, 0,
                                              a_f, (long long)NN * CC,
                                              nullptr, Abias, NN, DD, CC);
        k_proj<<<waves / 4, 128, 0, stream>>>(vT, (long long)NN * DD, Bw, 0,
                                              bm_f, (long long)NN * CC,
                                              nullptr, Bbias, NN, DD, CC);
    }
    {
        int waves = BB * (CC / 16) * (NN / 64);           // 4096
        k_proj<<<waves / 4, 128, 0, stream>>>(Gw, 0, vT, (long long)NN * DD,
                                              g_f, (long long)CC * NN,
                                              g_b, nullptr, CC, DD, NN);
    }

    // 4) fused R-GEMM + y-GEMM + W-GEMM (R never leaves the WGP)
    {
        int waves = BB * (NN / 16);                       // 1024
        k_fused<<<waves / 4, 128, 0, stream>>>(a_f, bm_f, g_f, Wwf, wy);
    }

    // 5) BN statistics (deterministic), 6) normalize + residual
    k_bnstats<<<dim3(DD / 64), dim3(64, 4), 0, stream>>>(wy, bnsum, bnsumsq);
    k_final<<<dim3(NN / 16, DD / 16, BB), dim3(16, 16), 0, stream>>>(
        wy, v, bnsum, bnsumsq, bn_gamma, bn_beta, (float*)d_out);
}
